// VisionModel_87729001988393
// MI455X (gfx1250) — matrix operand order repaired and checked by reading, not compiled, not run on hardware
//
#include <hip/hip_runtime.h>
#include <hip/hip_bf16.h>

// ---------------------------------------------------------------------------
// CDNA5 WMMA (wave32): D(16x16 f32) = A(16x32 bf16) x B(32x16 bf16) + C
// ---------------------------------------------------------------------------
typedef __attribute__((ext_vector_type(8)))  __bf16 v8bf;
typedef __attribute__((ext_vector_type(16))) __bf16 v16bf;
typedef __attribute__((ext_vector_type(8)))  float  v8f;

#define SEQ       2048
#define HIDDEN    1280
#define HEADS     16
#define HEAD_DIM  80
#define INTER     5120
#define DEPTH     4
#define PATCH_VOL 1536
#define OUT_HIDDEN 2048
#define WIN       64

// Async global->LDS (CDNA5). Guarded: fall back to regular loads if absent.
#if defined(__has_builtin)
#if __has_builtin(__builtin_amdgcn_global_load_async_to_lds_b128)
#define HAVE_ASYNC 1
#endif
#endif
#ifndef HAVE_ASYNC
#define HAVE_ASYNC 0
#endif

#define AS1 __attribute__((address_space(1)))
#define AS3 __attribute__((address_space(3)))

// Builtin expects pointers to GCC-style int 4-vectors (per clang diagnostic):
//   (v4i __device__ *src, v4i __shared__ *dst, imm offset, imm cpol)
typedef int v4i_vs __attribute__((vector_size(16)));

__device__ __forceinline__ void wait_async_zero() {
#if HAVE_ASYNC
#if __has_builtin(__builtin_amdgcn_s_wait_asynccnt)
  __builtin_amdgcn_s_wait_asynccnt(0);
#else
  asm volatile("s_wait_asynccnt 0x0" ::: "memory");
#endif
#endif
}

__device__ __forceinline__ unsigned short f2bf(float f) {
  // round-to-nearest-even f32 -> bf16
  unsigned u = __float_as_uint(f);
  unsigned r = u + 0x7FFFu + ((u >> 16) & 1u);
  return (unsigned short)(r >> 16);
}

__device__ __forceinline__ v8f zero8() {
  v8f z;
#pragma unroll
  for (int i = 0; i < 8; ++i) z[i] = 0.0f;
  return z;
}

// A fragment (16x32 bf16): half 0 holds K=0..7 & 16..23, half 1 K=8..15 & 24..31.
__device__ __forceinline__ v16bf frag_a(const unsigned short* p, unsigned half) {
  const v8bf a = *reinterpret_cast<const v8bf*>(p + (half << 3));
  const v8bf b = *reinterpret_cast<const v8bf*>(p + 16 + (half << 3));
  return __builtin_shufflevector(a, b, 0,1,2,3,4,5,6,7,8,9,10,11,12,13,14,15);
}
// B fragment (32x16 bf16): lane = column; half 0 K=0..7 & 16..23, half 1 K=8..15 & 24..31.
__device__ __forceinline__ v16bf frag_b(const unsigned short* p, unsigned half) {
  const v8bf a = *reinterpret_cast<const v8bf*>(p + (half << 3));
  const v8bf b = *reinterpret_cast<const v8bf*>(p + 16 + (half << 3));
  return __builtin_shufflevector(a, b, 0,1,2,3,4,5,6,7,8,9,10,11,12,13,14,15);
}
__device__ __forceinline__ v8f wmma_bf16(v16bf a, v16bf b, v8f c) {
  return __builtin_amdgcn_wmma_f32_16x16x32_bf16(false, a, false, b, (short)0, c,
                                                 false, false);
}

// ---------------------------------------------------------------------------
// Tiled GEMM: C[M,N] = epi(A_bf16[M,K] @ B_f32[K,N] + bias [, + R_f32])
// A staged via async global->LDS (raw bf16 copy); B converted f32->bf16 and
// transposed into LDS. Double-buffered. M%128==0, N%128==0, K%32==0.
// ---------------------------------------------------------------------------
#define BM 128
#define BN 128
#define BK 32
#define LDA 40  // ushort stride: 80B rows, 16B aligned, bank-friendly
#define LDB 40

enum { MODE_BIAS = 0, MODE_GELU_T = 1, MODE_GELU_E = 2, MODE_RES = 3 };

template <int MODE, int OUTBF>
__global__ __launch_bounds__(256) void gemm_kernel(
    const unsigned short* __restrict__ A, const float* __restrict__ B,
    const float* __restrict__ bias, const float* __restrict__ R,
    void* __restrict__ Cv, int M, int N, int K) {
  __shared__ unsigned short As[2][BM * LDA];
  __shared__ unsigned short Bs[2][BN * LDB];

  const int tid = threadIdx.x;
  const int bm = blockIdx.y, bn = blockIdx.x;
  const int w = __builtin_amdgcn_readfirstlane(tid >> 5);  // scalar wave id
  const unsigned lane = tid & 31u;
  const int wm = w >> 2, wn = w & 3;  // 2x4 wave grid -> 64x32 per wave
  const unsigned l16 = lane & 15u, half = lane >> 4;
  const int nk = K / BK;

  v8f acc[4][2];
#pragma unroll
  for (int i = 0; i < 4; ++i)
#pragma unroll
    for (int j = 0; j < 2; ++j) acc[i][j] = zero8();

  // Stage A tile (128 x 32 bf16 = 64B rows) as 512 x 16B segments.
  auto stageA = [&](int kt, int buf) {
    const int k0 = kt * BK;
#pragma unroll
    for (int p = 0; p < 2; ++p) {
      const int idx = p * 256 + tid;
      const int row = idx >> 2, seg = idx & 3;
      const unsigned short* src = A + (size_t)(bm * BM + row) * K + k0 + seg * 8;
      unsigned short* dst = &As[buf][row * LDA + seg * 8];
#if HAVE_ASYNC
      __builtin_amdgcn_global_load_async_to_lds_b128(
          (AS1 v4i_vs*)(unsigned long long)(uintptr_t)src,
          (AS3 v4i_vs*)(unsigned)(uintptr_t)dst, 0, 0);
#else
      *reinterpret_cast<uint4*>(dst) = *reinterpret_cast<const uint4*>(src);
#endif
    }
  };
  // Stage B tile (32 x 128 f32), convert + transpose -> Bs[n][k] bf16.
  auto stageB = [&](int kt, int buf) {
    const int k0 = kt * BK;
#pragma unroll
    for (int p = 0; p < 4; ++p) {
      const int e = p * 1024 + tid * 4;
      const int kk = e >> 7, nn = e & 127;
      const float4 v = *reinterpret_cast<const float4*>(
          B + (size_t)(k0 + kk) * N + bn * BN + nn);
      Bs[buf][(nn + 0) * LDB + kk] = f2bf(v.x);
      Bs[buf][(nn + 1) * LDB + kk] = f2bf(v.y);
      Bs[buf][(nn + 2) * LDB + kk] = f2bf(v.z);
      Bs[buf][(nn + 3) * LDB + kk] = f2bf(v.w);
    }
  };
  auto compute = [&](int buf) {
    v16bf af[4], bfr[2];
#pragma unroll
    for (int i = 0; i < 4; ++i)
      af[i] = frag_a(&As[buf][(wm * 64 + i * 16 + l16) * LDA], half);
#pragma unroll
    for (int j = 0; j < 2; ++j)
      bfr[j] = frag_b(&Bs[buf][(wn * 32 + j * 16 + l16) * LDB], half);
#pragma unroll
    for (int i = 0; i < 4; ++i)
#pragma unroll
      for (int j = 0; j < 2; ++j) acc[i][j] = wmma_bf16(af[i], bfr[j], acc[i][j]);
  };

  stageA(0, 0);
  stageB(0, 0);
  wait_async_zero();
  __syncthreads();
  for (int kt = 0; kt < nk; ++kt) {
    const int cur = kt & 1;
    if (kt + 1 < nk) {  // stage next tile into the other buffer
      stageA(kt + 1, cur ^ 1);
      stageB(kt + 1, cur ^ 1);
    }
    compute(cur);       // WMMA burst covers async latency
    wait_async_zero();
    __syncthreads();
  }

  // Epilogue: D VGPR r -> M = tile + half*8 + r, N = tile + l16
#pragma unroll
  for (int i = 0; i < 4; ++i) {
#pragma unroll
    for (int j = 0; j < 2; ++j) {
      const int c = bn * BN + wn * 32 + j * 16 + (int)l16;
      const int r0 = bm * BM + wm * 64 + i * 16 + (int)half * 8;
      const float bv = bias[c];
#pragma unroll
      for (int r = 0; r < 8; ++r) {
        float v = acc[i][j][r] + bv;
        if (MODE == MODE_GELU_T) {
          const float x3 = v * v * v;
          v = 0.5f * v * (1.0f + tanhf(0.7978845608028654f * (v + 0.044715f * x3)));
        } else if (MODE == MODE_GELU_E) {
          v = 0.5f * v * (1.0f + erff(v * 0.7071067811865476f));
        } else if (MODE == MODE_RES) {
          v += R[(size_t)(r0 + r) * N + c];
        }
        if (OUTBF)
          ((unsigned short*)Cv)[(size_t)(r0 + r) * N + c] = f2bf(v);
        else
          ((float*)Cv)[(size_t)(r0 + r) * N + c] = v;
      }
    }
  }
}

// ---------------------------------------------------------------------------
// LayerNorm (eps 1e-6), one block per row, bf16 output (feeds GEMM A operand)
// ---------------------------------------------------------------------------
__global__ __launch_bounds__(256) void ln_kernel(const float* __restrict__ x,
                                                 const float* __restrict__ s,
                                                 const float* __restrict__ b,
                                                 unsigned short* __restrict__ y,
                                                 int N) {
  __shared__ float rs[256];
  __shared__ float rq[256];
  const int row = blockIdx.x;
  const int tid = threadIdx.x;
  const float* xr = x + (size_t)row * N;
  float sum = 0.0f, sq = 0.0f;
  for (int i = tid; i < N; i += 256) {
    const float v = xr[i];
    sum += v;
    sq += v * v;
  }
  rs[tid] = sum;
  rq[tid] = sq;
  __syncthreads();
  for (int off = 128; off > 0; off >>= 1) {
    if (tid < off) {
      rs[tid] += rs[tid + off];
      rq[tid] += rq[tid + off];
    }
    __syncthreads();
  }
  const float mean = rs[0] / (float)N;
  const float var = rq[0] / (float)N - mean * mean;
  const float rstd = rsqrtf(var + 1e-6f);
  for (int i = tid; i < N; i += 256)
    y[(size_t)row * N + i] = f2bf((xr[i] - mean) * rstd * s[i] + b[i]);
}

// f32 -> bf16 elementwise (patch-embed input)
__global__ __launch_bounds__(256) void cvt_kernel(const float* __restrict__ in,
                                                  unsigned short* __restrict__ out,
                                                  int n) {
  const int i = blockIdx.x * 256 + threadIdx.x;
  if (i < n) out[i] = f2bf(in[i]);
}

// ---------------------------------------------------------------------------
// RoPE on q/k thirds of qkv (f32, in place). Pairs (d, d+40) per thread.
// ---------------------------------------------------------------------------
__global__ __launch_bounds__(256) void rope_kernel(float* __restrict__ qkv,
                                                   const float* __restrict__ cs,
                                                   const float* __restrict__ sn) {
  const int idx = blockIdx.x * 256 + threadIdx.x;
  const int t = idx / 1280;
  const int r = idx % 1280;
  const int tensor = r / 640;  // 0=q, 1=k
  const int r2 = r % 640;
  const int head = r2 / 40, p = r2 % 40;
  float* base = qkv + (size_t)t * (3 * HIDDEN) + tensor * HIDDEN + head * HEAD_DIM;
  const float c0 = cs[t * HEAD_DIM + p], s0 = sn[t * HEAD_DIM + p];
  const float c1 = cs[t * HEAD_DIM + p + 40], s1 = sn[t * HEAD_DIM + p + 40];
  const float a = base[p], bv = base[p + 40];
  base[p] = a * c0 - bv * s0;
  base[p + 40] = bv * c1 + a * s1;
}

// ---------------------------------------------------------------------------
// Windowed attention: one block per (window, head). WIN=64, HEAD_DIM=80.
// Intra-window mask is identically zero (cu_seqlens stride == WIN).
// Output bf16 (feeds out-proj GEMM A operand).
// ---------------------------------------------------------------------------
#define AQS 104  // Q/K LDS stride: 208B rows, 16B aligned
#define ASS 68   // scores stride (f32)
#define APS 72   // P/V LDS stride: 144B rows, 16B aligned

__global__ __launch_bounds__(256) void attn_kernel(const float* __restrict__ qkv,
                                                   unsigned short* __restrict__ o) {
  __shared__ unsigned short Qs[64 * AQS];
  __shared__ unsigned short Ks[64 * AQS];
  __shared__ float          Ss[64 * ASS];
  __shared__ unsigned short Ps[64 * APS];
  __shared__ unsigned short Vs[HEAD_DIM * APS];

  const int win = blockIdx.x;
  const int head = blockIdx.y;
  const int tid = threadIdx.x;
  const int t0 = win * WIN;
  const float scale = 0.11180339887498949f;  // 80^-0.5 folded into Q

  for (int idx = tid; idx < 64 * 96; idx += 256) {  // zero-pad K-dim to 96
    const int r = idx / 96, d = idx % 96;
    float qv = 0.0f, kv = 0.0f;
    if (d < HEAD_DIM) {
      const float* p = qkv + (size_t)(t0 + r) * (3 * HIDDEN) + head * HEAD_DIM + d;
      qv = p[0] * scale;
      kv = p[HIDDEN];
    }
    Qs[r * AQS + d] = f2bf(qv);
    Ks[r * AQS + d] = f2bf(kv);
  }
  for (int idx = tid; idx < 64 * HEAD_DIM; idx += 256) {  // Vs[dim][key]
    const int r = idx / HEAD_DIM, d = idx % HEAD_DIM;
    const float vv =
        qkv[(size_t)(t0 + r) * (3 * HIDDEN) + 2 * HIDDEN + head * HEAD_DIM + d];
    Vs[d * APS + r] = f2bf(vv);
  }
  __syncthreads();

  const int w = __builtin_amdgcn_readfirstlane(tid >> 5);  // scalar wave id
  const unsigned lane = tid & 31u;
  const unsigned l16 = lane & 15u, half = lane >> 4;

  // scores = Q K^T : 4x4 tile grid, wave w owns tiles w and w+8 (uniform flow)
  auto score_tile = [&](int tile) {
    const int tm = tile >> 2, tn = tile & 3;
    v8f acc = zero8();
#pragma unroll
    for (int ks = 0; ks < 3; ++ks) {
      const v16bf af = frag_a(&Qs[(tm * 16 + l16) * AQS + ks * 32], half);
      const v16bf bfr = frag_b(&Ks[(tn * 16 + l16) * AQS + ks * 32], half);
      acc = wmma_bf16(af, bfr, acc);
    }
#pragma unroll
    for (int r = 0; r < 8; ++r)
      Ss[(tm * 16 + (int)half * 8 + r) * ASS + tn * 16 + (int)l16] = acc[r];
  };
  score_tile(w);
  score_tile(w + 8);
  __syncthreads();

  // softmax per query row -> P (bf16, A-operand layout: contiguous keys)
  if (tid < 64) {
    const float* srow = &Ss[tid * ASS];
    float m = srow[0];
    for (int j = 1; j < 64; ++j) m = fmaxf(m, srow[j]);
    float sum = 0.0f;
    for (int j = 0; j < 64; ++j) sum += __expf(srow[j] - m);
    const float inv = 1.0f / sum;
    for (int j = 0; j < 64; ++j) Ps[tid * APS + j] = f2bf(__expf(srow[j] - m) * inv);
  }
  __syncthreads();

  // O = P V : 4x5 tile grid (20 tiles); wave w owns w, w+8, and w+16 if w<4
  auto pv_tile = [&](int tile) {
    const int tm = tile / 5, tn = tile % 5;
    v8f acc = zero8();
#pragma unroll
    for (int ks = 0; ks < 2; ++ks) {
      const v16bf af = frag_a(&Ps[(tm * 16 + l16) * APS + ks * 32], half);
      const v16bf bfr = frag_b(&Vs[(tn * 16 + l16) * APS + ks * 32], half);
      acc = wmma_bf16(af, bfr, acc);
    }
#pragma unroll
    for (int r = 0; r < 8; ++r) {
      const int q = tm * 16 + (int)half * 8 + r;
      const int d = tn * 16 + (int)l16;
      o[(size_t)(t0 + q) * HIDDEN + head * HEAD_DIM + d] = f2bf(acc[r]);
    }
  };
  pv_tile(w);
  pv_tile(w + 8);
  if (w < 4) pv_tile(w + 16);  // scalar branch: EXEC stays all-ones
}

// ---------------------------------------------------------------------------
// Host launcher
// ---------------------------------------------------------------------------
extern "C" void kernel_launch(void* const* d_in, const int* in_sizes, int n_in,
                              void* d_out, int out_size, void* d_ws, size_t ws_size,
                              hipStream_t stream) {
  (void)in_sizes; (void)n_in; (void)out_size; (void)ws_size;

  const float* x       = (const float*)d_in[0];
  const float* cosb    = (const float*)d_in[1];
  const float* sinb    = (const float*)d_in[2];
  /* d_in[3] = cu_seqlens (arange step WIN) — window size hardcoded */
  const float* patch_w = (const float*)d_in[4];
  const float* patch_b = (const float*)d_in[5];
  const float* ln1_s   = (const float*)d_in[6];
  const float* ln1_b   = (const float*)d_in[7];
  const float* qkv_w   = (const float*)d_in[8];
  const float* qkv_b   = (const float*)d_in[9];
  const float* out_w   = (const float*)d_in[10];
  const float* out_b   = (const float*)d_in[11];
  const float* ln2_s   = (const float*)d_in[12];
  const float* ln2_b   = (const float*)d_in[13];
  const float* fc1_w   = (const float*)d_in[14];
  const float* fc1_b   = (const float*)d_in[15];
  const float* fc2_w   = (const float*)d_in[16];
  const float* fc2_b   = (const float*)d_in[17];
  const float* mns     = (const float*)d_in[18];
  const float* mnb     = (const float*)d_in[19];
  const float* m1w     = (const float*)d_in[20];
  const float* m1b     = (const float*)d_in[21];
  const float* m2w     = (const float*)d_in[22];
  const float* m2b     = (const float*)d_in[23];

  // Workspace layout (~85 MB), all chunks 256B-aligned
  char* ws = (char*)d_ws;
  float* h            = (float*)ws;                 ws += (size_t)SEQ * HIDDEN * 4;
  float* qkv          = (float*)ws;                 ws += (size_t)SEQ * 3 * HIDDEN * 4;
  unsigned short* y   = (unsigned short*)ws;        ws += (size_t)SEQ * HIDDEN * 2;
  unsigned short* o   = (unsigned short*)ws;        ws += (size_t)SEQ * HIDDEN * 2;
  unsigned short* t   = (unsigned short*)ws;        ws += (size_t)SEQ * INTER * 2;
  unsigned short* t2  = (unsigned short*)ws;        ws += (size_t)(SEQ / 4) * INTER * 2;
  unsigned short* xb  = (unsigned short*)ws;        ws += (size_t)SEQ * PATCH_VOL * 2;

  const dim3 blk(256);

  // x -> bf16, then patch embed: h = xb @ patch_w + patch_b
  cvt_kernel<<<(SEQ * PATCH_VOL) / 256, blk, 0, stream>>>(x, xb, SEQ * PATCH_VOL);
  gemm_kernel<MODE_BIAS, 0><<<dim3(HIDDEN / BN, SEQ / BM), blk, 0, stream>>>(
      xb, patch_w, patch_b, nullptr, h, SEQ, HIDDEN, PATCH_VOL);

  for (int d = 0; d < DEPTH; ++d) {
    // attention
    ln_kernel<<<SEQ, blk, 0, stream>>>(h, ln1_s + d * HIDDEN, ln1_b + d * HIDDEN, y,
                                       HIDDEN);
    gemm_kernel<MODE_BIAS, 0><<<dim3(3 * HIDDEN / BN, SEQ / BM), blk, 0, stream>>>(
        y, qkv_w + (size_t)d * HIDDEN * 3 * HIDDEN, qkv_b + d * 3 * HIDDEN, nullptr,
        qkv, SEQ, 3 * HIDDEN, HIDDEN);
    rope_kernel<<<(SEQ * 1280) / 256, blk, 0, stream>>>(qkv, cosb, sinb);
    attn_kernel<<<dim3(SEQ / WIN, HEADS), blk, 0, stream>>>(qkv, o);
    gemm_kernel<MODE_RES, 0><<<dim3(HIDDEN / BN, SEQ / BM), blk, 0, stream>>>(
        o, out_w + (size_t)d * HIDDEN * HIDDEN, out_b + d * HIDDEN, h, h, SEQ,
        HIDDEN, HIDDEN);
    // mlp
    ln_kernel<<<SEQ, blk, 0, stream>>>(h, ln2_s + d * HIDDEN, ln2_b + d * HIDDEN, y,
                                       HIDDEN);
    gemm_kernel<MODE_GELU_T, 1><<<dim3(INTER / BN, SEQ / BM), blk, 0, stream>>>(
        y, fc1_w + (size_t)d * HIDDEN * INTER, fc1_b + d * INTER, nullptr, t, SEQ,
        INTER, HIDDEN);
    gemm_kernel<MODE_RES, 0><<<dim3(HIDDEN / BN, SEQ / BM), blk, 0, stream>>>(
        t, fc2_w + (size_t)d * INTER * HIDDEN, fc2_b + d * HIDDEN, h, h, SEQ, HIDDEN,
        INTER);
  }

  // merger: LN on [2048,1280] -> bf16, viewed as [512,5120]
  ln_kernel<<<SEQ, blk, 0, stream>>>(h, mns, mnb, y, HIDDEN);
  gemm_kernel<MODE_GELU_E, 1><<<dim3(INTER / BN, (SEQ / 4) / BM), blk, 0, stream>>>(
      y, m1w, m1b, nullptr, t2, SEQ / 4, INTER, INTER);
  gemm_kernel<MODE_BIAS, 0><<<dim3(OUT_HIDDEN / BN, (SEQ / 4) / BM), blk, 0, stream>>>(
      t2, m2w, m2b, nullptr, (float*)d_out, SEQ / 4, OUT_HIDDEN, INTER);
}
